// AMDNetTiny_34222299414668
// MI455X (gfx1250) — matrix operand order repaired and checked by reading, not compiled, not run on hardware
//
#include <hip/hip_runtime.h>

typedef __attribute__((ext_vector_type(16))) _Float16 v16h;
typedef __attribute__((ext_vector_type(8)))  _Float16 v8h;
typedef __attribute__((ext_vector_type(8)))  float    v8f;
typedef int v4i_vs __attribute__((vector_size(16)));   // matches builtin param

#define AS_GLOBAL __attribute__((address_space(1)))
#define AS_LDS    __attribute__((address_space(3)))

#if __has_builtin(__builtin_amdgcn_global_load_async_to_lds_b128) && \
    __has_builtin(__builtin_amdgcn_s_wait_asynccnt)
#define CDNA5_ASYNC_LDS 1
#endif

// ---------------------------------------------------------------------------
// Weight prep: OIHW f32 -> per-lane WMMA B-fragment layout, f16, zero padded.
// Fragment element (g, chunk, lane, e) holds B[k][n] with
//   k = chunk*32 + (e>>3)*16 + (lane>>4)*8 + (e&7)   (K = ky*3*CIN + kx*CIN + ci)
//   n = g*16 + (lane&15)
// ---------------------------------------------------------------------------
__global__ void prep_wfrag_kernel(const float* __restrict__ w,
                                  _Float16* __restrict__ dst,
                                  int CIN, int COUT, int KS, int total)
{
  const int K   = KS * KS * CIN;
  const int NCH = (K + 31) >> 5;
  for (int idx = blockIdx.x * blockDim.x + threadIdx.x; idx < total;
       idx += gridDim.x * blockDim.x) {
    int e    = idx & 15;
    int l    = (idx >> 4) & 31;
    int rest = idx >> 9;
    int c    = rest % NCH;
    int g    = rest / NCH;
    int k    = c * 32 + (e >> 3) * 16 + (l >> 4) * 8 + (e & 7);
    int o    = g * 16 + (l & 15);
    float v = 0.0f;
    if (k < K) {
      int tap = k / CIN;
      int ci  = k - tap * CIN;
      int ky  = tap / KS;
      int kx  = tap - ky * KS;
      v = w[((o * CIN + ci) * KS + ky) * KS + kx];
    }
    dst[idx] = (_Float16)v;
  }
}

// ---------------------------------------------------------------------------
// Pixel-unshuffle (2x): x (2,4,1024,1024) f32 NCHW -> NHWC f16 (2,512,512,16).
// One thread per output pixel: 8x float2 loads, one v16h (32B) store.
// ---------------------------------------------------------------------------
__global__ void unshuffle_kernel(const float* __restrict__ x,
                                 _Float16* __restrict__ a0)
{
  const int total = 2 * 512 * 512;
  int idx = blockIdx.x * blockDim.x + threadIdx.x;
  if (idx >= total) return;
  int px = idx & 511;
  int py = (idx >> 9) & 511;
  int n  = idx >> 18;
  union { v16h v; _Float16 h[16]; } o;
  #pragma unroll
  for (int ci = 0; ci < 4; ++ci) {
    #pragma unroll
    for (int i = 0; i < 2; ++i) {
      const float* row =
          x + ((size_t)(n * 4 + ci) * 1024 + (py * 2 + i)) * 1024 + px * 2;
      float2 p = *(const float2*)row;
      o.h[ci * 4 + i * 2 + 0] = (_Float16)p.x;
      o.h[ci * 4 + i * 2 + 1] = (_Float16)p.y;
    }
  }
  *(v16h*)(a0 + (size_t)idx * 16) = o.v;
}

// ---------------------------------------------------------------------------
// Implicit-GEMM conv + bias + ReLU with v_wmma_f32_16x16x32_f16.
// NHWC activations with arbitrary per-pixel channel stride (fused concat).
// 256-thread block (8 wave32) -> 8-row x 16-col output tile; wave w owns row
// y0+w. Patch staged to LDS via async global->LDS b128 when available.
// ---------------------------------------------------------------------------
template <int CIN, int COUT, int KS>
__global__ void __launch_bounds__(256)
conv_wmma_kernel(const _Float16* __restrict__ in, int in_cs,
                 const _Float16* __restrict__ wfrag,
                 const float* __restrict__ bias,
                 _Float16* __restrict__ out, int out_cs,
                 int H, int W)
{
  static_assert(CIN % 8 == 0 && COUT % 16 == 0, "channel layout");
  constexpr int PAD = (KS == 3) ? 1 : 0;
  constexpr int TR = 8, TC = 16;
  constexpr int PR = TR + 2 * PAD, PC = TC + 2 * PAD;
  constexpr int K   = KS * KS * CIN;      // multiple of 16 for all layers here
  constexpr int NCH = (K + 31) / 32;
  constexpr int NG  = COUT / 16;

  __shared__ _Float16 patch[PR][PC][CIN];

  const int n   = blockIdx.z;
  const int x0  = blockIdx.x * TC;
  const int y0  = blockIdx.y * TR;
  const int tid = threadIdx.x;

  constexpr int C8 = CIN / 8;
  constexpr int NCHUNK = PR * PC * C8;    // 16-byte transfer units

#ifdef CDNA5_ASYNC_LDS
  // Boundary workgroups: pre-zero the patch so skipped (OOB) async lanes
  // leave zero padding behind. Disjoint addresses from the async writes.
  if (KS == 3 && (x0 == 0 || y0 == 0 || x0 + TC == W || y0 + TR == H)) {
    for (int i = tid; i < NCHUNK; i += 256) {
      int c8 = i % C8;
      int pc = (i / C8) % PC;
      int pr = i / (C8 * PC);
      v8h z = {};
      *(v8h*)(&patch[pr][pc][c8 * 8]) = z;
    }
  }
  for (int i = tid; i < NCHUNK; i += 256) {
    int c8 = i % C8;
    int pc = (i / C8) % PC;
    int pr = i / (C8 * PC);
    int gy = y0 + pr - PAD;
    int gx = x0 + pc - PAD;
    if (gy >= 0 && gy < H && gx >= 0 && gx < W) {
      const _Float16* src =
          in + ((size_t)(n * H + gy) * W + gx) * in_cs + c8 * 8;
      __builtin_amdgcn_global_load_async_to_lds_b128(
          (AS_GLOBAL v4i_vs*)(void*)src,
          (AS_LDS v4i_vs*)(void*)(&patch[pr][pc][c8 * 8]), 0, 0);
    }
  }
  __builtin_amdgcn_s_wait_asynccnt(0);
  __syncthreads();
#else
  for (int i = tid; i < NCHUNK; i += 256) {
    int c8 = i % C8;
    int pc = (i / C8) % PC;
    int pr = i / (C8 * PC);
    int gy = y0 + pr - PAD;
    int gx = x0 + pc - PAD;
    v8h v = {};
    if (gy >= 0 && gy < H && gx >= 0 && gx < W) {
      v = *(const v8h*)(in + ((size_t)(n * H + gy) * W + gx) * in_cs + c8 * 8);
    }
    *(v8h*)(&patch[pr][pc][c8 * 8]) = v;
  }
  __syncthreads();
#endif

  const int lane  = tid & 31;
  const int wid   = tid >> 5;   // output row within tile
  const int mrow  = lane & 15;  // A-matrix row (pixel) for this lane
  const int khalf = lane >> 4;  // K phase of A/B fragments

  v8f acc[NG] = {};
  const v16h* __restrict__ wf = (const v16h*)wfrag;

  #pragma unroll
  for (int c = 0; c < NCH; ++c) {
    // A fragment: lane (l<16) holds M=l, K = {base..base+7, base+16..base+23}
    union { v16h v; v8h h[2]; } A;
    #pragma unroll
    for (int half = 0; half < 2; ++half) {
      const int kb = c * 32 + half * 16 + khalf * 8;
      v8h part = {};
      if (kb < K) {
        if constexpr (KS == 1) {
          part = *(const v8h*)(&patch[wid][mrow][kb]);
        } else {
          // 8-element K runs never cross a tap boundary (CIN % 8 == 0)
          const int tap = kb / CIN;
          const int ci  = kb - tap * CIN;
          const int ty  = tap / 3;
          const int tx  = tap - ty * 3;
          part = *(const v8h*)(&patch[wid + ty][mrow + tx][ci]);
        }
      }
      A.h[half] = part;
    }
    #pragma unroll
    for (int g = 0; g < NG; ++g) {
      v16h B = wf[(g * NCH + c) * 32 + lane];
      acc[g] = __builtin_amdgcn_wmma_f32_16x16x32_f16(
          false, A.v, false, B, (short)0, acc[g], false, false);
    }
  }

  // D layout: lane holds N = lane&15, VGPR r holds M = khalf*8 + r.
  const int col = lane & 15;
  const int y   = y0 + wid;
  #pragma unroll
  for (int g = 0; g < NG; ++g) {
    const float bv = bias[g * 16 + col];
    #pragma unroll
    for (int r = 0; r < 8; ++r) {
      const int px = x0 + khalf * 8 + r;
      float v = acc[g][r] + bv;
      v = v > 0.0f ? v : 0.0f;  // every U-Net block conv is ReLU
      out[((size_t)(n * H + y) * W + px) * out_cs + g * 16 + col] = (_Float16)v;
    }
  }
}

// ---------------------------------------------------------------------------
// 2x2 max pool, NHWC f16, vectorized 8 channels (b128) per thread.
// ---------------------------------------------------------------------------
__global__ void maxpool_kernel(const _Float16* __restrict__ in, int in_cs,
                               _Float16* __restrict__ out, int out_cs,
                               int C8cnt, int Ho, int Wo, int total)
{
  int idx = blockIdx.x * blockDim.x + threadIdx.x;
  if (idx >= total) return;
  int c8 = idx % C8cnt;
  int t  = idx / C8cnt;
  int x  = t % Wo; t /= Wo;
  int y  = t % Ho; t /= Ho;
  int n  = t;
  const int Hi = Ho * 2, Wi = Wo * 2;
  const _Float16* p = in + ((size_t)(n * Hi + y * 2) * Wi + x * 2) * in_cs + c8 * 8;
  v8h a = *(const v8h*)p;
  v8h b = *(const v8h*)(p + in_cs);
  v8h c = *(const v8h*)(p + (size_t)Wi * in_cs);
  v8h d = *(const v8h*)(p + (size_t)Wi * in_cs + in_cs);
  v8h r;
  #pragma unroll
  for (int i = 0; i < 8; ++i) {
    float m = fmaxf(fmaxf((float)a[i], (float)b[i]),
                    fmaxf((float)c[i], (float)d[i]));
    r[i] = (_Float16)m;
  }
  *(v8h*)(out + ((size_t)(n * Ho + y) * Wo + x) * out_cs + c8 * 8) = r;
}

// ---------------------------------------------------------------------------
// Nearest 2x upsample into the concat buffer (channel offset folded into
// `out`, wide channel stride). 8 channels (b128) per thread.
// ---------------------------------------------------------------------------
__global__ void upsample_kernel(const _Float16* __restrict__ in, int in_cs,
                                _Float16* __restrict__ out, int out_cs,
                                int C8cnt, int Hi, int Wi, int total)
{
  int idx = blockIdx.x * blockDim.x + threadIdx.x;
  if (idx >= total) return;
  const int Ho = Hi * 2, Wo = Wi * 2;
  int c8 = idx % C8cnt;
  int t  = idx / C8cnt;
  int x  = t % Wo; t /= Wo;
  int y  = t % Ho; t /= Ho;
  int n  = t;
  v8h v = *(const v8h*)(in +
      ((size_t)(n * Hi + (y >> 1)) * Wi + (x >> 1)) * in_cs + c8 * 8);
  *(v8h*)(out + ((size_t)(n * Ho + y) * Wo + x) * out_cs + c8 * 8) = v;
}

// ---------------------------------------------------------------------------
// Fused tail: implicit pixel-shuffle read of unet output (32ch @512^2 NHWC),
// 1x1 pred conv 8->11, softmax(10) + sigmoid, dynamic 3x3 filter on sr_color
// + history blend. Writes final (2,3,1024,1024) then recurrent (2,1,1024,1024).
// ---------------------------------------------------------------------------
__global__ void final_kernel(const _Float16* __restrict__ uout,
                             const float* __restrict__ sr,
                             const float* __restrict__ prev,
                             const float* __restrict__ pw,
                             const float* __restrict__ pb,
                             float* __restrict__ out)
{
  const int H = 1024, W = 1024, N = 2;
  int idx = blockIdx.x * blockDim.x + threadIdx.x;
  if (idx >= N * H * W) return;
  int x = idx % W;
  int y = (idx / W) % H;
  int n = idx / (W * H);

  // pixel_shuffle2: feats[n,c,y,x] = unet[n, y>>1, x>>1, c*4 + (y&1)*2+(x&1)]
  const int off = (y & 1) * 2 + (x & 1);
  const _Float16* up =
      uout + ((size_t)(n * 512 + (y >> 1)) * 512 + (x >> 1)) * 32 + off;
  float f[8];
  #pragma unroll
  for (int c = 0; c < 8; ++c) f[c] = (float)up[c * 4];

  float p[11];
  #pragma unroll
  for (int k = 0; k < 11; ++k) {
    float s = pb[k];
    #pragma unroll
    for (int c = 0; c < 8; ++c) s += pw[k * 8 + c] * f[c];
    p[k] = s;
  }

  float m = p[0];
  #pragma unroll
  for (int k = 1; k < 10; ++k) m = fmaxf(m, p[k]);
  float e[10], sum = 0.0f;
  #pragma unroll
  for (int k = 0; k < 10; ++k) { e[k] = __expf(p[k] - m); sum += e[k]; }
  const float inv  = 1.0f / sum;
  const float hist = e[9] * inv;
  const float rec  = 1.0f / (1.0f + __expf(-p[10]));

  #pragma unroll
  for (int c = 0; c < 3; ++c) {
    const float* img = sr + (size_t)(n * 3 + c) * H * W;
    float acc = 0.0f;
    #pragma unroll
    for (int k = 0; k < 9; ++k) {
      int dy = k / 3 - 1, dx = k % 3 - 1;
      int yy = y + dy, xx = x + dx;
      float pv = (yy >= 0 && yy < H && xx >= 0 && xx < W)
                     ? img[(size_t)yy * W + xx] : 0.0f;
      acc += pv * (e[k] * inv);
    }
    acc += prev[((size_t)(n * 3 + c) * H + y) * W + x] * hist;
    out[((size_t)(n * 3 + c) * H + y) * W + x] = acc;
  }
  out[(size_t)N * 3 * H * W + ((size_t)n * H + y) * W + x] = rec;
}

// ---------------------------------------------------------------------------
// Host orchestration
// ---------------------------------------------------------------------------
extern "C" void kernel_launch(void* const* d_in, const int* in_sizes, int n_in,
                              void* d_out, int out_size, void* d_ws,
                              size_t ws_size, hipStream_t stream)
{
  (void)in_sizes; (void)n_in; (void)out_size; (void)ws_size;

  // Inputs (params flattened in sorted-key order: bott,d1,d2,d3,e1,e2,e3,pred)
  const float* xin = (const float*)d_in[0];
  const float* sr  = (const float*)d_in[1];
  const float* pv  = (const float*)d_in[2];
  const float* Wb1 = (const float*)d_in[3];  const float* Bb1 = (const float*)d_in[4];
  const float* Wb2 = (const float*)d_in[5];  const float* Bb2 = (const float*)d_in[6];
  const float* Wd1a= (const float*)d_in[7];  const float* Bd1a= (const float*)d_in[8];
  const float* Wd1b= (const float*)d_in[9];  const float* Bd1b= (const float*)d_in[10];
  const float* Wd2a= (const float*)d_in[11]; const float* Bd2a= (const float*)d_in[12];
  const float* Wd2b= (const float*)d_in[13]; const float* Bd2b= (const float*)d_in[14];
  const float* Wd3a= (const float*)d_in[15]; const float* Bd3a= (const float*)d_in[16];
  const float* Wd3b= (const float*)d_in[17]; const float* Bd3b= (const float*)d_in[18];
  const float* We1a= (const float*)d_in[19]; const float* Be1a= (const float*)d_in[20];
  const float* We1b= (const float*)d_in[21]; const float* Be1b= (const float*)d_in[22];
  const float* We2a= (const float*)d_in[23]; const float* Be2a= (const float*)d_in[24];
  const float* We2b= (const float*)d_in[25]; const float* Be2b= (const float*)d_in[26];
  const float* We3a= (const float*)d_in[27]; const float* Be3a= (const float*)d_in[28];
  const float* We3b= (const float*)d_in[29]; const float* Be3b= (const float*)d_in[30];
  const float* Wpr = (const float*)d_in[31]; const float* Bpr = (const float*)d_in[32];

  char* ws = (char*)d_ws;
  size_t off = 0;
  auto alloc = [&](size_t halves) {
    size_t o = off;
    off += (halves * 2 + 255) & ~(size_t)255;
    return o;
  };
  auto F16 = [&](size_t o) { return (_Float16*)(ws + o); };

  // Activation buffers (NHWC f16; concat buffers hold skips at channel offset)
  const size_t a0   = alloc(2ull * 512 * 512 * 16);
  const size_t t1   = alloc(2ull * 512 * 512 * 16);
  const size_t d1in = alloc(2ull * 512 * 512 * 48);   // [u1 | skip1@32]
  const size_t p1   = alloc(2ull * 256 * 256 * 16);
  const size_t t2   = alloc(2ull * 256 * 256 * 32);
  const size_t d2in = alloc(2ull * 256 * 256 * 80);   // [u2 | skip2@48]
  const size_t p2   = alloc(2ull * 128 * 128 * 32);
  const size_t t3   = alloc(2ull * 128 * 128 * 48);
  const size_t d3in = alloc(2ull * 128 * 128 * 112);  // [u3 | skip3@64]
  const size_t p3   = alloc(2ull * 64 * 64 * 48);
  const size_t tb   = alloc(2ull * 64 * 64 * 64);
  const size_t bo   = alloc(2ull * 64 * 64 * 64);
  const size_t d3m  = alloc(2ull * 128 * 128 * 48);
  const size_t d3o  = alloc(2ull * 128 * 128 * 48);
  const size_t d2m  = alloc(2ull * 256 * 256 * 32);
  const size_t d2o  = alloc(2ull * 256 * 256 * 32);
  const size_t d1m  = alloc(2ull * 512 * 512 * 32);
  const size_t uo   = alloc(2ull * 512 * 512 * 32);

  // Weight fragment buffers
  auto nch = [](int cin, int ks) { return (ks * ks * cin + 31) / 32; };
  auto fragHalves = [&](int cin, int cout, int ks) {
    return (size_t)(cout / 16) * nch(cin, ks) * 512;
  };
  struct L { const float* w; int cin, cout, ks; size_t o; };
  L ls[14] = {
    {We1a, 16, 16, 3, 0}, {We1b, 16, 16, 3, 0},
    {We2a, 16, 32, 3, 0}, {We2b, 32, 32, 3, 0},
    {We3a, 32, 48, 3, 0}, {We3b, 48, 48, 3, 0},
    {Wb1,  48, 64, 3, 0}, {Wb2,  64, 64, 3, 0},
    {Wd3a, 112,48, 1, 0}, {Wd3b, 48, 48, 3, 0},
    {Wd2a, 80, 32, 1, 0}, {Wd2b, 32, 32, 3, 0},
    {Wd1a, 48, 32, 1, 0}, {Wd1b, 32, 32, 3, 0},
  };
  for (int i = 0; i < 14; ++i) ls[i].o = alloc(fragHalves(ls[i].cin, ls[i].cout, ls[i].ks));
  for (int i = 0; i < 14; ++i) {
    int total = (int)fragHalves(ls[i].cin, ls[i].cout, ls[i].ks);
    prep_wfrag_kernel<<<(total + 255) / 256, 256, 0, stream>>>(
        ls[i].w, F16(ls[i].o), ls[i].cin, ls[i].cout, ls[i].ks, total);
  }

  auto cgrid = [](int H, int W) { return dim3(W / 16, H / 8, 2); };

  // ---- encoder ----
  unshuffle_kernel<<<(2 * 512 * 512 + 255) / 256, 256, 0, stream>>>(xin, F16(a0));

  conv_wmma_kernel<16, 16, 3><<<cgrid(512, 512), 256, 0, stream>>>(
      F16(a0), 16, F16(ls[0].o), Be1a, F16(t1), 16, 512, 512);
  conv_wmma_kernel<16, 16, 3><<<cgrid(512, 512), 256, 0, stream>>>(
      F16(t1), 16, F16(ls[1].o), Be1b, F16(d1in) + 32, 48, 512, 512);  // skip1

  {
    int total = 2 * 256 * 256 * 2;
    maxpool_kernel<<<(total + 255) / 256, 256, 0, stream>>>(
        F16(d1in) + 32, 48, F16(p1), 16, 2, 256, 256, total);
  }
  conv_wmma_kernel<16, 32, 3><<<cgrid(256, 256), 256, 0, stream>>>(
      F16(p1), 16, F16(ls[2].o), Be2a, F16(t2), 32, 256, 256);
  conv_wmma_kernel<32, 32, 3><<<cgrid(256, 256), 256, 0, stream>>>(
      F16(t2), 32, F16(ls[3].o), Be2b, F16(d2in) + 48, 80, 256, 256);  // skip2

  {
    int total = 2 * 128 * 128 * 4;
    maxpool_kernel<<<(total + 255) / 256, 256, 0, stream>>>(
        F16(d2in) + 48, 80, F16(p2), 32, 4, 128, 128, total);
  }
  conv_wmma_kernel<32, 48, 3><<<cgrid(128, 128), 256, 0, stream>>>(
      F16(p2), 32, F16(ls[4].o), Be3a, F16(t3), 48, 128, 128);
  conv_wmma_kernel<48, 48, 3><<<cgrid(128, 128), 256, 0, stream>>>(
      F16(t3), 48, F16(ls[5].o), Be3b, F16(d3in) + 64, 112, 128, 128); // skip3

  // ---- bottleneck ----
  {
    int total = 2 * 64 * 64 * 6;
    maxpool_kernel<<<(total + 255) / 256, 256, 0, stream>>>(
        F16(d3in) + 64, 112, F16(p3), 48, 6, 64, 64, total);
  }
  conv_wmma_kernel<48, 64, 3><<<cgrid(64, 64), 256, 0, stream>>>(
      F16(p3), 48, F16(ls[6].o), Bb1, F16(tb), 64, 64, 64);
  conv_wmma_kernel<64, 64, 3><<<cgrid(64, 64), 256, 0, stream>>>(
      F16(tb), 64, F16(ls[7].o), Bb2, F16(bo), 64, 64, 64);

  // ---- decoder ----
  {
    int total = 2 * 128 * 128 * 8;
    upsample_kernel<<<(total + 255) / 256, 256, 0, stream>>>(
        F16(bo), 64, F16(d3in), 112, 8, 64, 64, total);
  }
  conv_wmma_kernel<112, 48, 1><<<cgrid(128, 128), 256, 0, stream>>>(
      F16(d3in), 112, F16(ls[8].o), Bd3a, F16(d3m), 48, 128, 128);
  conv_wmma_kernel<48, 48, 3><<<cgrid(128, 128), 256, 0, stream>>>(
      F16(d3m), 48, F16(ls[9].o), Bd3b, F16(d3o), 48, 128, 128);

  {
    int total = 2 * 256 * 256 * 6;
    upsample_kernel<<<(total + 255) / 256, 256, 0, stream>>>(
        F16(d3o), 48, F16(d2in), 80, 6, 128, 128, total);
  }
  conv_wmma_kernel<80, 32, 1><<<cgrid(256, 256), 256, 0, stream>>>(
      F16(d2in), 80, F16(ls[10].o), Bd2a, F16(d2m), 32, 256, 256);
  conv_wmma_kernel<32, 32, 3><<<cgrid(256, 256), 256, 0, stream>>>(
      F16(d2m), 32, F16(ls[11].o), Bd2b, F16(d2o), 32, 256, 256);

  {
    int total = 2 * 512 * 512 * 4;
    upsample_kernel<<<(total + 255) / 256, 256, 0, stream>>>(
        F16(d2o), 32, F16(d1in), 48, 4, 256, 256, total);
  }
  conv_wmma_kernel<48, 32, 1><<<cgrid(512, 512), 256, 0, stream>>>(
      F16(d1in), 48, F16(ls[12].o), Bd1a, F16(d1m), 32, 512, 512);
  conv_wmma_kernel<32, 32, 3><<<cgrid(512, 512), 256, 0, stream>>>(
      F16(d1m), 32, F16(ls[13].o), Bd1b, F16(uo), 32, 512, 512);

  // ---- fused pred + softmax + dynamic filter + history blend ----
  final_kernel<<<(2 * 1024 * 1024 + 255) / 256, 256, 0, stream>>>(
      F16(uo), sr, pv, Wpr, Bpr, (float*)d_out);
}
